// MoE_66803921322559
// MI455X (gfx1250) — matrix operand order repaired and checked by reading, not compiled, not run on hardware
//
#include <hip/hip_runtime.h>
#include <hip/hip_bf16.h>

typedef __bf16 bf16;
typedef __attribute__((ext_vector_type(16))) __bf16 v16bf;
typedef __attribute__((ext_vector_type(8)))  float  v8f;
typedef int v2i __attribute__((vector_size(8)));

#define T_TOK   2048
#define DIMD    1024
#define NEXP    16
#define TOPK    2
#define INTER_R 512
#define INTER_S 1024

#define AS1 __attribute__((address_space(1)))
#define AS3 __attribute__((address_space(3)))

#if defined(__has_builtin)
#if __has_builtin(__builtin_amdgcn_global_load_async_to_lds_b64)
#define HAVE_ASYNC 1
#endif
#endif
#ifndef HAVE_ASYNC
#define HAVE_ASYNC 0
#endif

// ---------- scalar helpers ----------
__device__ inline bf16 f2bf(float f) { return (bf16)f; }   // native v_cvt_*_bf16_f32

__device__ inline float fast_rcp(float x) {
#if __has_builtin(__builtin_amdgcn_rcpf)
    return __builtin_amdgcn_rcpf(x);                       // v_rcp_f32
#else
    return 1.f / x;
#endif
}
__device__ inline float silu(float g) { return g * fast_rcp(1.f + __expf(-g)); }

__device__ inline void async_wait0() {
#if HAVE_ASYNC
#if __has_builtin(__builtin_amdgcn_s_wait_asynccnt)
    __builtin_amdgcn_s_wait_asynccnt(0);
#else
    asm volatile("s_wait_asynccnt 0x0" ::: "memory");
#endif
#endif
}

// ---------- WMMA fragment helpers (CDNA5 wave32 layouts) ----------
// A (16x32 bf16, MxK) read from an LDS tile laid out [32 rows][32 k] bf16:
// lanes 0-15: row=lane,   K {0..7, 16..23};  lanes 16-31: row=lane-16, K {8..15, 24..31}
__device__ inline v16bf lds_a_frag(const bf16* __restrict__ smem, int lane, int wrow) {
    const bf16* p = smem + ((wrow << 4) + (lane & 15)) * 32 + (lane >> 4) * 8;
    v16bf a;
#pragma unroll
    for (int i = 0; i < 8; i++) { a[i] = p[i]; a[i + 8] = p[i + 16]; }
    return a;
}

// B (32x16 bf16, KxN) from fp32 weight stored [N, K] row-major:
// lanes 0-15: N=n, K {0..7, 16..23}; lanes 16-31: N=n, K {8..15, 24..31}
__device__ inline v16bf load_b_f32(const float* __restrict__ w, int ldk, int n, int k0, int lane) {
    const float* p = w + (size_t)n * ldk + k0 + (lane >> 4) * 8;
    v16bf b;
#pragma unroll
    for (int i = 0; i < 8; i++) { b[i] = f2bf(p[i]); b[i + 8] = f2bf(p[i + 16]); }
    return b;
}

__device__ inline v8f wmma_bf16(v16bf a, v16bf b, v8f c) {
    return __builtin_amdgcn_wmma_f32_16x16x32_bf16(false, a, false, b, (short)0, c, false, false);
}

// ---------- A-tile staging: 32 rows x 32 k (bf16) = 2 KB, 256 threads x 8 B ----------
__device__ inline void stage_a_tile(const bf16* __restrict__ base, int ld,
                                    const int* __restrict__ rows_s, int k0,
                                    bf16* __restrict__ dst, int tid) {
    int row = tid >> 3, q = tid & 7;
    const bf16* g = base + (size_t)rows_s[row] * ld + k0 + q * 4;
    bf16* l = dst + row * 32 + q * 4;
#if HAVE_ASYNC
    __builtin_amdgcn_global_load_async_to_lds_b64((AS1 v2i*)g, (AS3 v2i*)l, 0, 0);
#else
    unsigned long long v;
    __builtin_memcpy(&v, g, 8);
    __builtin_memcpy(l, &v, 8);
#endif
}

// ---------- 1) gate: scores, top-2, x -> bf16 ----------
__global__ void moe_gate_kernel(const float* __restrict__ x, const float* __restrict__ gw,
                                bf16* __restrict__ xb, int* __restrict__ topi,
                                float* __restrict__ topw, int* __restrict__ counts) {
    __shared__ float part[64];
    __shared__ float score[NEXP];
    int t = blockIdx.x, tid = threadIdx.x;           // 64 threads / token
    const float* xr = x + (size_t)t * DIMD;
#pragma unroll
    for (int i = 0; i < 16; i++) {
        int k = tid * 16 + i;
        xb[(size_t)t * DIMD + k] = f2bf(xr[k]);
    }
    int e = tid & 15, seg = tid >> 4;                // 16 experts x 4 segments of D
    const float* wr = gw + (size_t)e * DIMD + seg * 256;
    const float* xs = xr + seg * 256;
    float s = 0.f;
    for (int k = 0; k < 256; k++) s += xs[k] * wr[k];
    part[tid] = s;
    __syncthreads();
    if (tid < NEXP) {
        float sc = part[tid] + part[tid + 16] + part[tid + 32] + part[tid + 48];
        score[tid] = fast_rcp(1.f + __expf(-sc));
    }
    __syncthreads();
    if (tid == 0) {
        int i0 = 0; float v0 = score[0];
        for (int i = 1; i < NEXP; i++) if (score[i] > v0) { v0 = score[i]; i0 = i; }
        int i1 = (i0 == 0) ? 1 : 0; float v1 = score[i1];
        for (int i = 0; i < NEXP; i++) if (i != i0 && score[i] > v1) { v1 = score[i]; i1 = i; }
        float inv = 1.f / (v0 + v1);
        topi[t * 2] = i0; topi[t * 2 + 1] = i1;
        topw[t * 2] = v0 * inv; topw[t * 2 + 1] = v1 * inv;
        atomicAdd(&counts[i0], 1); atomicAdd(&counts[i1], 1);
    }
}

// ---------- 2) routing bookkeeping ----------
__global__ void moe_zero_kernel(int* __restrict__ counts) {
    if (threadIdx.x < NEXP) counts[threadIdx.x] = 0;
}

__global__ void moe_scan_kernel(const int* __restrict__ counts, int* __restrict__ offs,
                                int* __restrict__ cursor) {
    if (threadIdx.x == 0) {
        int acc = 0;
        for (int e = 0; e < NEXP; e++) { offs[e] = acc; cursor[e] = acc; acc += counts[e]; }
    }
}

__global__ void moe_scatter_kernel(const int* __restrict__ topi, const float* __restrict__ topw,
                                   int* __restrict__ cursor, int* __restrict__ slot_tok,
                                   float* __restrict__ slot_wt) {
    int t = blockIdx.x * blockDim.x + threadIdx.x;
    if (t >= T_TOK) return;
#pragma unroll
    for (int j = 0; j < TOPK; j++) {
        int e = topi[t * 2 + j];
        int pos = atomicAdd(&cursor[e], 1);
        slot_tok[pos] = t;
        slot_wt[pos] = topw[t * 2 + j];
    }
}

// ---------- 3) shared up: hs = silu(x@sw1^T) * (x@sw3^T). block 32M x 128N ----------
__global__ void moe_shared_up_kernel(const bf16* __restrict__ xb, const float* __restrict__ sw1,
                                     const float* __restrict__ sw3, bf16* __restrict__ hs) {
    __shared__ __attribute__((aligned(16))) bf16 atile[2][32 * 32];
    __shared__ int rows_s[32];
    int tid = threadIdx.x, lane = tid & 31, w = tid >> 5;
    int wrow = w >> 2;
    int mb = blockIdx.x * 32;
    int n0 = blockIdx.y * 128 + (w & 3) * 32;
    if (tid < 32) rows_s[tid] = mb + tid;
    __syncthreads();
    stage_a_tile(xb, DIMD, rows_s, 0, atile[0], tid);
    int nA = n0 + (lane & 15), nB = nA + 16;
    v8f c1a = {}, c1b = {}, c3a = {}, c3b = {};
    int buf = 0;
    for (int k0 = 0; k0 < DIMD; k0 += 32, buf ^= 1) {
        async_wait0();
        __syncthreads();
        if (k0 + 32 < DIMD) stage_a_tile(xb, DIMD, rows_s, k0 + 32, atile[buf ^ 1], tid);
        __builtin_prefetch(sw1 + (size_t)nA * DIMD + k0 + 256, 0, 0);
        __builtin_prefetch(sw3 + (size_t)nA * DIMD + k0 + 256, 0, 0);
        v16bf a   = lds_a_frag(atile[buf], lane, wrow);
        v16bf b1a = load_b_f32(sw1, DIMD, nA, k0, lane);
        v16bf b1b = load_b_f32(sw1, DIMD, nB, k0, lane);
        v16bf b3a = load_b_f32(sw3, DIMD, nA, k0, lane);
        v16bf b3b = load_b_f32(sw3, DIMD, nB, k0, lane);
        c1a = wmma_bf16(a, b1a, c1a);
        c1b = wmma_bf16(a, b1b, c1b);
        c3a = wmma_bf16(a, b3a, c3a);
        c3b = wmma_bf16(a, b3b, c3b);
        __syncthreads();
    }
    int m0 = mb + wrow * 16 + (lane >> 4) * 8;
#pragma unroll
    for (int r = 0; r < 8; r++) {
        int m = m0 + r;
        hs[(size_t)m * INTER_S + nA] = f2bf(silu(c1a[r]) * c3a[r]);
        hs[(size_t)m * INTER_S + nB] = f2bf(silu(c1b[r]) * c3b[r]);
    }
}

// ---------- 4) shared down: out = hs @ sw2^T (initializes d_out) ----------
__global__ void moe_shared_down_kernel(const bf16* __restrict__ hs, const float* __restrict__ sw2,
                                       float* __restrict__ out) {
    __shared__ __attribute__((aligned(16))) bf16 atile[2][32 * 32];
    __shared__ int rows_s[32];
    int tid = threadIdx.x, lane = tid & 31, w = tid >> 5;
    int wrow = w >> 2;
    int mb = blockIdx.x * 32;
    int n0 = blockIdx.y * 128 + (w & 3) * 32;
    if (tid < 32) rows_s[tid] = mb + tid;
    __syncthreads();
    stage_a_tile(hs, INTER_S, rows_s, 0, atile[0], tid);
    int nA = n0 + (lane & 15), nB = nA + 16;
    v8f ca = {}, cb = {};
    int buf = 0;
    for (int k0 = 0; k0 < INTER_S; k0 += 32, buf ^= 1) {
        async_wait0();
        __syncthreads();
        if (k0 + 32 < INTER_S) stage_a_tile(hs, INTER_S, rows_s, k0 + 32, atile[buf ^ 1], tid);
        __builtin_prefetch(sw2 + (size_t)nA * INTER_S + k0 + 256, 0, 0);
        v16bf a  = lds_a_frag(atile[buf], lane, wrow);
        v16bf ba = load_b_f32(sw2, INTER_S, nA, k0, lane);
        v16bf bb = load_b_f32(sw2, INTER_S, nB, k0, lane);
        ca = wmma_bf16(a, ba, ca);
        cb = wmma_bf16(a, bb, cb);
        __syncthreads();
    }
    int m0 = mb + wrow * 16 + (lane >> 4) * 8;
#pragma unroll
    for (int r = 0; r < 8; r++) {
        out[(size_t)(m0 + r) * DIMD + nA] = ca[r];
        out[(size_t)(m0 + r) * DIMD + nB] = cb[r];
    }
}

// ---------- 5) routed up: per-expert dense GEMM over gathered tokens ----------
__global__ void moe_routed_up_kernel(const bf16* __restrict__ xb, const float* __restrict__ w1,
                                     const float* __restrict__ w3, const int* __restrict__ offs,
                                     const int* __restrict__ counts, const int* __restrict__ slot_tok,
                                     bf16* __restrict__ hr) {
    __shared__ __attribute__((aligned(16))) bf16 atile[2][32 * 32];
    __shared__ int rows_s[32];
    int e = blockIdx.z;
    int cnt = counts[e];
    int mbase = blockIdx.x * 32;
    if (mbase >= cnt) return;                 // block-uniform exit (EXEC stays full for WMMA)
    int off = offs[e];
    int tid = threadIdx.x, lane = tid & 31, w = tid >> 5;
    int wrow = w >> 2;
    int n0 = blockIdx.y * 128 + (w & 3) * 32;
    if (tid < 32) rows_s[tid] = slot_tok[off + min(mbase + tid, cnt - 1)];
    __syncthreads();
    stage_a_tile(xb, DIMD, rows_s, 0, atile[0], tid);
    const float* w1e = w1 + (size_t)e * INTER_R * DIMD;
    const float* w3e = w3 + (size_t)e * INTER_R * DIMD;
    int nA = n0 + (lane & 15), nB = nA + 16;
    v8f c1a = {}, c1b = {}, c3a = {}, c3b = {};
    int buf = 0;
    for (int k0 = 0; k0 < DIMD; k0 += 32, buf ^= 1) {
        async_wait0();
        __syncthreads();
        if (k0 + 32 < DIMD) stage_a_tile(xb, DIMD, rows_s, k0 + 32, atile[buf ^ 1], tid);
        __builtin_prefetch(w1e + (size_t)nA * DIMD + k0 + 256, 0, 0);
        __builtin_prefetch(w3e + (size_t)nA * DIMD + k0 + 256, 0, 0);
        v16bf a   = lds_a_frag(atile[buf], lane, wrow);
        v16bf b1a = load_b_f32(w1e, DIMD, nA, k0, lane);
        v16bf b1b = load_b_f32(w1e, DIMD, nB, k0, lane);
        v16bf b3a = load_b_f32(w3e, DIMD, nA, k0, lane);
        v16bf b3b = load_b_f32(w3e, DIMD, nB, k0, lane);
        c1a = wmma_bf16(a, b1a, c1a);
        c1b = wmma_bf16(a, b1b, c1b);
        c3a = wmma_bf16(a, b3a, c3a);
        c3b = wmma_bf16(a, b3b, c3b);
        __syncthreads();
    }
    int m0 = mbase + wrow * 16 + (lane >> 4) * 8;
#pragma unroll
    for (int r = 0; r < 8; r++) {
        int m = m0 + r;
        if (m < cnt) {
            hr[(size_t)(off + m) * INTER_R + nA] = f2bf(silu(c1a[r]) * c3a[r]);
            hr[(size_t)(off + m) * INTER_R + nB] = f2bf(silu(c1b[r]) * c3b[r]);
        }
    }
}

// ---------- 6) routed down: contrib = hr @ w2e^T, weighted atomic scatter ----------
__global__ void moe_routed_down_kernel(const bf16* __restrict__ hr, const float* __restrict__ w2,
                                       const int* __restrict__ offs, const int* __restrict__ counts,
                                       const int* __restrict__ slot_tok, const float* __restrict__ slot_wt,
                                       float* __restrict__ out) {
    __shared__ __attribute__((aligned(16))) bf16 atile[2][32 * 32];
    __shared__ int rows_s[32];
    int e = blockIdx.z;
    int cnt = counts[e];
    int mbase = blockIdx.x * 32;
    if (mbase >= cnt) return;
    int off = offs[e];
    int tid = threadIdx.x, lane = tid & 31, w = tid >> 5;
    int wrow = w >> 2;
    int n0 = blockIdx.y * 128 + (w & 3) * 32;
    if (tid < 32) rows_s[tid] = off + min(mbase + tid, cnt - 1);
    __syncthreads();
    stage_a_tile(hr, INTER_R, rows_s, 0, atile[0], tid);
    const float* w2e = w2 + (size_t)e * DIMD * INTER_R;
    int nA = n0 + (lane & 15), nB = nA + 16;
    v8f ca = {}, cb = {};
    int buf = 0;
    for (int k0 = 0; k0 < INTER_R; k0 += 32, buf ^= 1) {
        async_wait0();
        __syncthreads();
        if (k0 + 32 < INTER_R) stage_a_tile(hr, INTER_R, rows_s, k0 + 32, atile[buf ^ 1], tid);
        __builtin_prefetch(w2e + (size_t)nA * INTER_R + k0 + 256, 0, 0);
        v16bf a  = lds_a_frag(atile[buf], lane, wrow);
        v16bf ba = load_b_f32(w2e, INTER_R, nA, k0, lane);
        v16bf bb = load_b_f32(w2e, INTER_R, nB, k0, lane);
        ca = wmma_bf16(a, ba, ca);
        cb = wmma_bf16(a, bb, cb);
        __syncthreads();
    }
    int m0 = mbase + wrow * 16 + (lane >> 4) * 8;
#pragma unroll
    for (int r = 0; r < 8; r++) {
        int m = m0 + r;
        if (m < cnt) {
            int tok = slot_tok[off + m];
            float wt = slot_wt[off + m];
            atomicAdd(&out[(size_t)tok * DIMD + nA], wt * ca[r]);
            atomicAdd(&out[(size_t)tok * DIMD + nB], wt * cb[r]);
        }
    }
}

// ---------- launch ----------
extern "C" void kernel_launch(void* const* d_in, const int* in_sizes, int n_in,
                              void* d_out, int out_size, void* d_ws, size_t ws_size,
                              hipStream_t stream) {
    const float* x   = (const float*)d_in[0];
    const float* gw  = (const float*)d_in[1];
    const float* w1  = (const float*)d_in[2];
    const float* w2  = (const float*)d_in[3];
    const float* w3  = (const float*)d_in[4];
    const float* sw1 = (const float*)d_in[5];
    const float* sw2 = (const float*)d_in[6];
    const float* sw3 = (const float*)d_in[7];
    float* out = (float*)d_out;

    char* ws = (char*)d_ws;
    bf16*  xb       = (bf16*)ws;  ws += (size_t)T_TOK * DIMD * sizeof(bf16);
    bf16*  hs       = (bf16*)ws;  ws += (size_t)T_TOK * INTER_S * sizeof(bf16);
    bf16*  hr       = (bf16*)ws;  ws += (size_t)T_TOK * TOPK * INTER_R * sizeof(bf16);
    int*   counts   = (int*)ws;   ws += NEXP * sizeof(int);
    int*   offs     = (int*)ws;   ws += NEXP * sizeof(int);
    int*   cursor   = (int*)ws;   ws += NEXP * sizeof(int);
    int*   topi     = (int*)ws;   ws += (size_t)T_TOK * TOPK * sizeof(int);
    float* topw     = (float*)ws; ws += (size_t)T_TOK * TOPK * sizeof(float);
    int*   slot_tok = (int*)ws;   ws += (size_t)T_TOK * TOPK * sizeof(int);
    float* slot_wt  = (float*)ws; ws += (size_t)T_TOK * TOPK * sizeof(float);

    moe_zero_kernel<<<dim3(1), dim3(32), 0, stream>>>(counts);
    moe_gate_kernel<<<dim3(T_TOK), dim3(64), 0, stream>>>(x, gw, xb, topi, topw, counts);
    moe_scan_kernel<<<dim3(1), dim3(32), 0, stream>>>(counts, offs, cursor);
    moe_scatter_kernel<<<dim3(T_TOK / 256), dim3(256), 0, stream>>>(topi, topw, cursor, slot_tok, slot_wt);

    moe_shared_up_kernel<<<dim3(T_TOK / 32, INTER_S / 128), dim3(256), 0, stream>>>(xb, sw1, sw3, hs);
    moe_shared_down_kernel<<<dim3(T_TOK / 32, DIMD / 128), dim3(256), 0, stream>>>(hs, sw2, out);

    moe_routed_up_kernel<<<dim3(T_TOK * TOPK / 32, INTER_R / 128, NEXP), dim3(256), 0, stream>>>(
        xb, w1, w3, offs, counts, slot_tok, hr);
    moe_routed_down_kernel<<<dim3(T_TOK * TOPK / 32, DIMD / 128, NEXP), dim3(256), 0, stream>>>(
        hr, w2, offs, counts, slot_tok, slot_wt, out);
}
